// ldmCrossAttnWG_77558519431684
// MI455X (gfx1250) — matrix operand order repaired and checked by reading, not compiled, not run on hardware
//
#include <hip/hip_runtime.h>

// ---------------------------------------------------------------------------
// Fused LDM cross-attention for MI455X (gfx1250), wave32 + WMMA f16.
//   sizes: B=16 NQ=4096 NKV=77 QD=320 CD=768 H=8 DH=64 INNER=512
// ~55 GFLOP, ~168MB mandatory HBM traffic (~7us @23.3TB/s).
// v_wmma_f32_16x16x32_f16 everywhere; register-blocked N-tile groups with
// batched fragment loads (one wait -> WMMA burst) and unroll-2 k-loops so the
// scheduler can pipeline loads of step k+1 over WMMAs of step k.
// ---------------------------------------------------------------------------

#define BSZ   16
#define NQ    4096
#define NKV   77
#define QD    320
#define CD    768
#define HEADS 8
#define DH    64
#define INNER 512
#define KVP   80      // kctx rows padded (5 N-tiles of 16)
#define KVP2  96      // V^T cols padded (3 K-steps of 32)
#define SIMLD 100     // sim row stride in floats (96 used + pad, breaks bank phase)
#define TQ    64      // query rows per block

#define SCALE   0.125f            // 64^-0.5
#define MAXNEG  (-1.70141173e38f) // -finfo(f32).max / 2

typedef __attribute__((ext_vector_type(16))) _Float16 v16h;
typedef __attribute__((ext_vector_type(8)))  _Float16 hv8;
typedef __attribute__((ext_vector_type(4)))  _Float16 hv4;
typedef __attribute__((ext_vector_type(8)))  float    v8f;
typedef __attribute__((ext_vector_type(4)))  float    fv4;

#define WMMA_F16(acc, a, b) \
  (acc) = __builtin_amdgcn_wmma_f32_16x16x32_f16(false, (a), false, (b), (short)0, (acc), false, false)

// ---- workspace layout (bytes) ----
static constexpr size_t OFF_WQT  = 0;                                   // 512x320 f16
static constexpr size_t OFF_WKT  = OFF_WQT + (size_t)INNER*QD*2;        // 512x768 f16
static constexpr size_t OFF_WVT  = OFF_WKT + (size_t)INNER*CD*2;        // 512x768 f16
static constexpr size_t OFF_WOT  = OFF_WVT + (size_t)INNER*CD*2;        // 320x512 f16
static constexpr size_t OFF_KCTX = OFF_WOT + (size_t)QD*INNER*2;        // 16x80x512 f16
static constexpr size_t OFF_VT   = OFF_KCTX + (size_t)BSZ*KVP*INNER*2;  // 16x8x64x96 f16

// ---- LDS layout for attn_main (bytes) ----
static constexpr int SM_QH   = 0;                           // 64x512 f16 = 65536 (reused as outA f16)
static constexpr int SM_SIM  = SM_QH  + TQ*INNER*2;         // 8x64x100 f32 = 204800
static constexpr int SM_XA   = SM_SIM + HEADS*TQ*SIMLD*4;   // 64x320 f16 = 40960
static constexpr int SM_RCP  = SM_XA  + TQ*QD*2;            // 512 f32
static constexpr int SM_MB   = SM_RCP + HEADS*TQ*4;         // 96 f32
static constexpr int SMEM_MAIN = SM_MB + 96*4;              // = 313,728 B  (<= 320KB WGP LDS)
static constexpr int SMEM_KV   = TQ*CD*2;                   // 98,304 B

// ---------------------------------------------------------------------------
// WMMA fragment loaders (per ISA 7.12.2 VGPR layouts, wave32)
// A 16x32 f16: lane l -> row m = l&15; halves 0..7 : k = k0 + (l>>4)*8 + {0..7}
//                                      halves 8..15: k = k0 + 16 + (l>>4)*8 + {0..7}
// B 32x16 f16: lane l -> col n = l&15; halves 0..7 : k = k0 + (l>>4)*8 + {0..7}
//                                      halves 8..15: k = k0 + 16 + (l>>4)*8 + {0..7}
// C/D 16x16 f32: lane l -> col n = l&15, VGPR j -> row m = 8*(l>>4)+j
// ---------------------------------------------------------------------------
__device__ __forceinline__ v16h frag_a(const _Float16* base, int row0, int ld, int k0, int lane) {
  const int g = lane >> 4, r = lane & 15;
  const _Float16* p = base + (size_t)(row0 + r) * ld + k0 + g * 8;
  hv8 lo = *(const hv8*)(p);
  hv8 hi = *(const hv8*)(p + 16);
  v16h v;
#pragma unroll
  for (int i = 0; i < 8; ++i) { v[i] = lo[i]; v[i + 8] = hi[i]; }
  return v;
}

__device__ __forceinline__ v16h frag_b(const _Float16* base, int col0, int ld, int k0, int lane) {
  const int g = lane >> 4, n = lane & 15;
  const _Float16* p = base + (size_t)(col0 + n) * ld + k0 + g * 8;
  hv8 lo = *(const hv8*)(p);
  hv8 hi = *(const hv8*)(p + 16);
  v16h v;
#pragma unroll
  for (int i = 0; i < 8; ++i) { v[i] = lo[i]; v[i + 8] = hi[i]; }
  return v;
}

// A-fragment built from f32 exp(sim) in LDS, folding 1/rowsum + f32->f16 convert.
__device__ __forceinline__ v16h frag_attn(const float* sh, int m0, int k0,
                                          const float* rowrcp, int lane) {
  const int g = lane >> 4, r = lane & 15;
  const int row = m0 + r;
  const float s = rowrcp[row];
  const float* p = sh + row * SIMLD + k0 + g * 8;
  fv4 a0 = *(const fv4*)(p);
  fv4 a1 = *(const fv4*)(p + 4);
  fv4 b0 = *(const fv4*)(p + 16);
  fv4 b1 = *(const fv4*)(p + 20);
  v16h v;
#pragma unroll
  for (int u = 0; u < 4; ++u) {
    v[u]      = (_Float16)(a0[u] * s);
    v[u + 4]  = (_Float16)(a1[u] * s);
    v[u + 8]  = (_Float16)(b0[u] * s);
    v[u + 12] = (_Float16)(b1[u] * s);
  }
  return v;
}

// ---------------------------------------------------------------------------
// Kernel 1: weight transpose/convert to f16 (NxK, K contiguous) + zero pads.
// ---------------------------------------------------------------------------
__global__ void prep_kernel(const float* __restrict__ Wq, const float* __restrict__ Wk,
                            const float* __restrict__ Wv, const float* __restrict__ Wo,
                            _Float16* __restrict__ WqT, _Float16* __restrict__ WkT,
                            _Float16* __restrict__ WvT, _Float16* __restrict__ WoT,
                            _Float16* __restrict__ kctx, _Float16* __restrict__ Vt) {
  const int tid = blockIdx.x * blockDim.x + threadIdx.x;
  const int stride = gridDim.x * blockDim.x;
  for (int i = tid; i < INNER * QD; i += stride) {           // WqT[n][k] = Wq[k][n]
    int n = i / QD, k = i % QD;
    WqT[i] = (_Float16)Wq[(size_t)k * INNER + n];
  }
  for (int i = tid; i < INNER * CD; i += stride) {           // WkT / WvT
    int n = i / CD, k = i % CD;
    WkT[i] = (_Float16)Wk[(size_t)k * INNER + n];
    WvT[i] = (_Float16)Wv[(size_t)k * INNER + n];
  }
  for (int i = tid; i < QD * INNER; i += stride) {           // WoT[n][k] = Wo[k][n]
    int n = i / INNER, k = i % INNER;
    WoT[i] = (_Float16)Wo[(size_t)k * QD + n];
  }
  for (int i = tid; i < BSZ * KVP * INNER; i += stride) kctx[i] = (_Float16)0.f;
  for (int i = tid; i < BSZ * HEADS * DH * KVP2; i += stride) Vt[i] = (_Float16)0.f;
}

// ---------------------------------------------------------------------------
// Kernel 2: K/V projection as one (16*77)x768 @ 768x512 WMMA GEMM.
// Writes kctx [b][kv(pad80)][512] and transposed Vt [b][h][d][kv(pad96)].
// grid (20, 4): x = 64-row M chunk, y = 8 N-tile chunk. 128 threads = 4 waves.
// ---------------------------------------------------------------------------
__global__ void __launch_bounds__(128, 1)
kv_proj_kernel(const float* __restrict__ ctx,
               const _Float16* __restrict__ WkT,
               const _Float16* __restrict__ WvT,
               _Float16* __restrict__ kctx,
               _Float16* __restrict__ Vt) {
  extern __shared__ char smem_raw[];
  _Float16* cA = (_Float16*)smem_raw;                 // 64 x 768 f16
  const int tid  = threadIdx.x;
  const int lane = tid & 31;
  const int wv   = tid >> 5;                          // 0..3 -> M-tile
  const int r0   = blockIdx.x * TQ;
  const int ROWS = BSZ * NKV;                         // 1232 = 77 * 16

  // load context tile -> f16 LDS (zero tail rows)
  for (int i = tid; i < TQ * (CD / 4); i += blockDim.x) {
    int r = i / (CD / 4), c4 = i % (CD / 4);
    int R = r0 + r;
    hv4 hv = {};
    if (R < ROWS) {
      fv4 xv = *(const fv4*)(ctx + (size_t)R * CD + c4 * 4);
#pragma unroll
      for (int u = 0; u < 4; ++u) hv[u] = (_Float16)xv[u];
    }
    *(hv4*)(cA + r * CD + c4 * 4) = hv;
  }
  __syncthreads();

  // M-tile validity is uniform per wave (ROWS is a multiple of 16).
  const bool mvalid = (r0 + wv * 16) < ROWS;
  if (!mvalid) return;

  const int nt0 = blockIdx.y * 8;
#pragma unroll 1
  for (int sel = 0; sel < 2; ++sel) {
    const _Float16* W = sel ? WvT : WkT;
#pragma unroll 1
    for (int ntg = 0; ntg < 2; ++ntg) {               // 2 groups of 4 N-tiles
      const int ntb = nt0 + ntg * 4;
      v8f acc[4] = {};
#pragma unroll 2
      for (int ks = 0; ks < CD / 32; ++ks) {
        v16h bb[4];
#pragma unroll
        for (int g = 0; g < 4; ++g) bb[g] = frag_b(W, (ntb + g) * 16, CD, ks * 32, lane);
        const v16h a = frag_a(cA, wv * 16, CD, ks * 32, lane);
#pragma unroll
        for (int g = 0; g < 4; ++g) WMMA_F16(acc[g], a, bb[g]);
      }
      const int gg = lane >> 4, col = lane & 15;
#pragma unroll
      for (int g = 0; g < 4; ++g) {
        const int n = (ntb + g) * 16 + col;
#pragma unroll
        for (int j = 0; j < 8; ++j) {
          int R = r0 + wv * 16 + gg * 8 + j;          // always < ROWS here
          int bb_ = R / NKV, kv = R - bb_ * NKV;
          if (sel == 0) {
            kctx[((size_t)(bb_ * KVP) + kv) * INNER + n] = (_Float16)acc[g][j];
          } else {
            int h = n >> 6, d = n & 63;
            Vt[((size_t)((bb_ * HEADS + h) * DH) + d) * KVP2 + kv] = (_Float16)acc[g][j];
          }
        }
      }
    }
  }
}

// ---------------------------------------------------------------------------
// Kernel 3: fused Qproj + attention + output projection.
// grid (NQ/64, B); 256 threads = 8 waves; wave == head in attention phases.
// ---------------------------------------------------------------------------
__global__ void __launch_bounds__(256, 1)
attn_main_kernel(const float* __restrict__ x,
                 const unsigned char* __restrict__ mask,
                 const float* __restrict__ bo,
                 const _Float16* __restrict__ WqT,
                 const _Float16* __restrict__ WoT,
                 const _Float16* __restrict__ kctx,
                 const _Float16* __restrict__ Vt,
                 float* __restrict__ out) {
  extern __shared__ char smem_raw[];
  _Float16* qh     = (_Float16*)(smem_raw + SM_QH);   // q f16 (phase1-2); outA f16 (phase4-5)
  float*    sim    = (float*)   (smem_raw + SM_SIM);  // 8 x 64 x SIMLD f32
  _Float16* xA     = (_Float16*)(smem_raw + SM_XA);   // 64 x 320 f16
  float*    rowrcp = (float*)   (smem_raw + SM_RCP);  // 8*64 f32
  float*    mb     = (float*)   (smem_raw + SM_MB);   // 96 f32 mask bias

  const int b    = blockIdx.y;
  const int q0   = blockIdx.x * TQ;
  const int tid  = threadIdx.x;
  const int lane = tid & 31;
  const int wv   = tid >> 5;                          // 0..7

  // ---- phase 1a: stage x tile as f16; build additive mask bias ----
  {
    const float* xb = x + (size_t)(b * NQ + q0) * QD;
    for (int i = tid; i < TQ * (QD / 4); i += 256) {
      int r = i / (QD / 4), c4 = i % (QD / 4);
      fv4 xv = *(const fv4*)(xb + (size_t)r * QD + c4 * 4);
      hv4 hv;
#pragma unroll
      for (int u = 0; u < 4; ++u) hv[u] = (_Float16)xv[u];
      *(hv4*)(xA + r * QD + c4 * 4) = hv;
    }
    if (tid < 96) mb[tid] = (tid < NKV && mask[b * NKV + tid]) ? 0.f : MAXNEG;
  }
  __syncthreads();

  // ---- phase 1b: q = x @ Wq (scale folded into f16 store) ----
  {
    const int m  = wv >> 1;
    const int n0 = (wv & 1) * 16;
#pragma unroll 1
    for (int ntg = 0; ntg < 4; ++ntg) {               // 4 groups of 4 N-tiles
      const int ntb = n0 + ntg * 4;
      v8f acc[4] = {};
#pragma unroll 2
      for (int ks = 0; ks < QD / 32; ++ks) {
        v16h bb[4];
#pragma unroll
        for (int g = 0; g < 4; ++g) bb[g] = frag_b(WqT, (ntb + g) * 16, QD, ks * 32, lane);
        const v16h a = frag_a(xA, m * 16, QD, ks * 32, lane);
#pragma unroll
        for (int g = 0; g < 4; ++g) WMMA_F16(acc[g], a, bb[g]);
      }
      const int gg = lane >> 4, col = lane & 15;
#pragma unroll
      for (int g = 0; g < 4; ++g)
#pragma unroll
        for (int j = 0; j < 8; ++j) {
          int row = m * 16 + gg * 8 + j;
          qh[row * INNER + (ntb + g) * 16 + col] = (_Float16)(acc[g][j] * SCALE);
        }
    }
  }
  __syncthreads();

  // ---- phase 2: sim = q @ K^T for own head (wave == head) ----
  {
    const int h = wv;
    const _Float16* kb = kctx + (size_t)b * KVP * INNER + h * DH;  // row=kv, K=dim contiguous
    float* sh = sim + (size_t)h * TQ * SIMLD;
#pragma unroll 1
    for (int mt = 0; mt < 4; ++mt) {
      v8f acc[5] = {};
#pragma unroll
      for (int ks = 0; ks < 2; ++ks) {
        v16h bb[5];
#pragma unroll
        for (int nt = 0; nt < 5; ++nt) bb[nt] = frag_b(kb, nt * 16, INNER, ks * 32, lane);
        const v16h a = frag_a(qh, mt * 16, INNER, h * DH + ks * 32, lane);
#pragma unroll
        for (int nt = 0; nt < 5; ++nt) WMMA_F16(acc[nt], a, bb[nt]);
      }
      const int gg = lane >> 4, col = lane & 15;
#pragma unroll
      for (int nt = 0; nt < 5; ++nt)
#pragma unroll
        for (int j = 0; j < 8; ++j) {
          int row = mt * 16 + gg * 8 + j;
          sh[row * SIMLD + nt * 16 + col] = acc[nt][j];
        }
    }
  }
  // own-wave producer/consumer on sim & qh columns of this head: no barrier needed.

  // ---- phase 3: masked softmax in place (store exp, keep 1/sum) ----
  {
    const int h = wv;
    float* sh = sim + (size_t)h * TQ * SIMLD;
#pragma unroll
    for (int rr = 0; rr < 2; ++rr) {
      const int row = lane + rr * 32;
      float* p = sh + row * SIMLD;
      float mx = -3.4e38f;
      for (int c4 = 0; c4 < 20; ++c4) {
        fv4 v = *(const fv4*)(p + c4 * 4);
        fv4 bv = *(const fv4*)(mb + c4 * 4);
        v += bv;
        mx = fmaxf(mx, fmaxf(fmaxf(v[0], v[1]), fmaxf(v[2], v[3])));
      }
      float s = 0.f;
      for (int c4 = 0; c4 < 20; ++c4) {
        fv4 v = *(const fv4*)(p + c4 * 4);
        fv4 bv = *(const fv4*)(mb + c4 * 4);
        fv4 e;
#pragma unroll
        for (int u = 0; u < 4; ++u) e[u] = __expf(v[u] + bv[u] - mx);
        s += e[0] + e[1] + e[2] + e[3];
        *(fv4*)(p + c4 * 4) = e;
      }
      rowrcp[h * TQ + row] = 1.f / s;
      fv4 z = {};   // zero pad cols 80..95 (3rd K-step of attn@V)
      *(fv4*)(p + 80) = z; *(fv4*)(p + 84) = z; *(fv4*)(p + 88) = z; *(fv4*)(p + 92) = z;
    }
  }

  // ---- phase 4: outA = attn @ V (A built on the fly from exp/sim) ----
  {
    const int h = wv;
    const float* sh = sim + (size_t)h * TQ * SIMLD;
    const float* rr = rowrcp + h * TQ;
    const _Float16* vb = Vt + (size_t)((b * HEADS + h) * DH) * KVP2;  // row=d, K=kv contiguous
#pragma unroll 1
    for (int mt = 0; mt < 4; ++mt) {
      v8f acc[4] = {};
#pragma unroll 1
      for (int ks = 0; ks < 3; ++ks) {
        v16h bb[4];
#pragma unroll
        for (int nt = 0; nt < 4; ++nt) bb[nt] = frag_b(vb, nt * 16, KVP2, ks * 32, lane);
        const v16h a = frag_attn(sh, mt * 16, ks * 32, rr, lane);
#pragma unroll
        for (int nt = 0; nt < 4; ++nt) WMMA_F16(acc[nt], a, bb[nt]);
      }
      const int gg = lane >> 4, col = lane & 15;
#pragma unroll
      for (int nt = 0; nt < 4; ++nt)
#pragma unroll
        for (int j = 0; j < 8; ++j) {
          int row = mt * 16 + gg * 8 + j;
          // reuse qh slab as f16 A for output projection (own head's columns only)
          qh[row * INNER + h * DH + nt * 16 + col] = (_Float16)acc[nt][j];
        }
    }
  }
  __syncthreads();

  // ---- phase 5: out = outA @ Wo + bo ----
  {
    const int m  = wv >> 1;
    const int n0 = (wv & 1) * 10;
    float* ob = out + (size_t)(b * NQ + q0) * QD;
#pragma unroll 1
    for (int ntg = 0; ntg < 2; ++ntg) {               // 2 groups of 5 N-tiles
      const int ntb = n0 + ntg * 5;
      v8f acc[5] = {};
#pragma unroll 2
      for (int ks = 0; ks < INNER / 32; ++ks) {
        v16h bb[5];
#pragma unroll
        for (int g = 0; g < 5; ++g) bb[g] = frag_b(WoT, (ntb + g) * 16, INNER, ks * 32, lane);
        const v16h a = frag_a(qh, m * 16, INNER, ks * 32, lane);
#pragma unroll
        for (int g = 0; g < 5; ++g) WMMA_F16(acc[g], a, bb[g]);
      }
      const int gg = lane >> 4, col = lane & 15;
#pragma unroll
      for (int g = 0; g < 5; ++g) {
        const int n = (ntb + g) * 16 + col;
        const float bias = bo[n];
#pragma unroll
        for (int j = 0; j < 8; ++j) {
          int row = m * 16 + gg * 8 + j;
          ob[(size_t)row * QD + n] = acc[g][j] + bias;
        }
      }
    }
  }
}

// ---------------------------------------------------------------------------
extern "C" void kernel_launch(void* const* d_in, const int* in_sizes, int n_in,
                              void* d_out, int out_size, void* d_ws, size_t ws_size,
                              hipStream_t stream) {
  (void)in_sizes; (void)n_in; (void)out_size; (void)ws_size;
  const float* x       = (const float*)d_in[0];
  const float* context = (const float*)d_in[1];
  const unsigned char* mask = (const unsigned char*)d_in[2];  // numpy bool = 1 byte
  const float* Wq = (const float*)d_in[3];
  const float* Wk = (const float*)d_in[4];
  const float* Wv = (const float*)d_in[5];
  const float* Wo = (const float*)d_in[6];
  const float* bo = (const float*)d_in[7];
  float* out = (float*)d_out;

  char* ws = (char*)d_ws;
  _Float16* WqT  = (_Float16*)(ws + OFF_WQT);
  _Float16* WkT  = (_Float16*)(ws + OFF_WKT);
  _Float16* WvT  = (_Float16*)(ws + OFF_WVT);
  _Float16* WoT  = (_Float16*)(ws + OFF_WOT);
  _Float16* kctx = (_Float16*)(ws + OFF_KCTX);
  _Float16* Vt   = (_Float16*)(ws + OFF_VT);

  (void)hipFuncSetAttribute(reinterpret_cast<const void*>(kv_proj_kernel),
                            hipFuncAttributeMaxDynamicSharedMemorySize, SMEM_KV);
  (void)hipFuncSetAttribute(reinterpret_cast<const void*>(attn_main_kernel),
                            hipFuncAttributeMaxDynamicSharedMemorySize, SMEM_MAIN);

  prep_kernel<<<512, 256, 0, stream>>>(Wq, Wk, Wv, Wo, WqT, WkT, WvT, WoT, kctx, Vt);
  kv_proj_kernel<<<dim3(20, 4), 128, SMEM_KV, stream>>>(context, WkT, WvT, kctx, Vt);
  attn_main_kernel<<<dim3(NQ / TQ, BSZ), 256, SMEM_MAIN, stream>>>(
      x, mask, bo, WqT, WoT, kctx, Vt, out);
}
